// gauss_MultiheadAttention_24713241822144
// MI455X (gfx1250) — matrix operand order repaired and checked by reading, not compiled, not run on hardware
//
#include <hip/hip_runtime.h>

typedef __bf16 bf16;
typedef __attribute__((ext_vector_type(16))) __bf16 v16bf;
typedef __attribute__((ext_vector_type(8)))  __bf16 v8bf;
typedef __attribute__((ext_vector_type(8)))  float  v8f;
typedef __attribute__((ext_vector_type(4)))  float  v4f;

static constexpr int Bn = 4, Ln = 2048, En = 1024, Hn = 16, HDn = 64, BHn = 64, Mn = Bn * Ln;
static constexpr float SIGMA = 10.0f, ATTN_BIAS = 0.1f, QSCALE = 0.125f; // 1/sqrt(64)

__device__ __forceinline__ v8f wmma_bf16(v16bf a, v16bf b, v8f c) {
  return __builtin_amdgcn_wmma_f32_16x16x32_bf16(false, a, false, b, (short)0, c, false, false);
}

__device__ __forceinline__ v16bf cat_bf(v8bf lo, v8bf hi) {
  v16bf r;
#pragma unroll
  for (int i = 0; i < 8; ++i) { r[i] = lo[i]; r[i + 8] = hi[i]; }
  return r;
}

// A fragment (16x32 bf16): lane = row lane%16; chunks at K=8*(lane/16), +16
__device__ __forceinline__ v16bf load_afrag(const bf16* arow, int k, int hsel) {
  v8bf lo = *(const v8bf*)(arow + k + hsel * 8);
  v8bf hi = *(const v8bf*)(arow + k + 16 + hsel * 8);
  return cat_bf(lo, hi);
}
// B fragment (32x16 bf16): lane = col lane%16; chunks at K=8*(lane/16), +16
__device__ __forceinline__ v16bf load_bfrag(const bf16* wrow, int k, int hsel) {
  return cat_bf(*(const v8bf*)(wrow + k + hsel * 8), *(const v8bf*)(wrow + k + 16 + hsel * 8));
}

// ---------------------------------------------------------------- prep kernels
__global__ void gw_kernel(const float* __restrict__ g, float* __restrict__ o) {
  int i = blockIdx.x * blockDim.x + threadIdx.x;
  if (i < (BHn / 2) * Ln) o[i] = __expf(-g[i] * SIGMA);
}

__global__ void cvt_kernel(const float* __restrict__ x, bf16* __restrict__ y, int n8) {
  int i = blockIdx.x * blockDim.x + threadIdx.x;
  if (i >= n8) return;
  const v4f* p = (const v4f*)(x + (size_t)i * 8);
  v4f a = p[0], b = p[1];
  v8bf r;
#pragma unroll
  for (int j = 0; j < 4; ++j) { r[j] = (__bf16)a[j]; r[j + 4] = (__bf16)b[j]; }
  *(v8bf*)(y + (size_t)i * 8) = r;
}

// ---------------------------------------------------------------- in-projection
// 32x64 tile per wave; k-loop unrolled x2 with ping-pong fragment buffers
// (no register rotation copies). mode 0:Q scaled; 1:K; 2:V transposed.
__global__ __launch_bounds__(128) void proj_kernel(
    const bf16* __restrict__ X, const bf16* __restrict__ W,
    const float* __restrict__ bias, bf16* __restrict__ out, int mode) {
  const int lane = threadIdx.x & 31;
  const int wv   = threadIdx.x >> 5;
  const int row  = lane & 15;
  const int hsel = lane >> 4;
  const int m0 = blockIdx.x * 128 + wv * 32;
  const int n0 = blockIdx.y * 64;

  const bf16* arow0 = X + (size_t)(m0 + row) * En;
  const bf16* arow1 = arow0 + (size_t)16 * En;
  const bf16* wrow[4];
#pragma unroll
  for (int t = 0; t < 4; ++t) wrow[t] = W + (size_t)(n0 + t * 16 + row) * En;

  v8f acc0[4] = {v8f{}, v8f{}, v8f{}, v8f{}};
  v8f acc1[4] = {v8f{}, v8f{}, v8f{}, v8f{}};

  // ping-pong fragment buffers
  v16bf a0A, a1A, bA[4], a0B, a1B, bB[4];

  auto loadset = [&](int k, v16bf& a0d, v16bf& a1d, v16bf (&bd)[4]) {
    a0d = load_afrag(arow0, k, hsel);
    a1d = load_afrag(arow1, k, hsel);
#pragma unroll
    for (int t = 0; t < 4; ++t) bd[t] = load_bfrag(wrow[t], k, hsel);
  };
  auto compute = [&](v16bf& a0c, v16bf& a1c, v16bf (&bc)[4]) {
#pragma unroll
    for (int t = 0; t < 4; ++t) {
      acc0[t] = wmma_bf16(a0c, bc[t], acc0[t]);
      acc1[t] = wmma_bf16(a1c, bc[t], acc1[t]);
    }
  };

  loadset(0, a0A, a1A, bA);
  for (int k = 0; k < En; k += 64) {
    loadset(k + 32, a0B, a1B, bB);               // load k+32 while computing k
    compute(a0A, a1A, bA);
    const int kn = (k + 64 < En) ? k + 64 : 0;   // harmless dummy reload at end
    loadset(kn, a0A, a1A, bA);                   // load k+64 while computing k+32
    compute(a0B, a1B, bB);
  }

  const int bidx = m0 >> 11;
#pragma unroll
  for (int mi = 0; mi < 2; ++mi) {
    const int l0 = ((m0 + mi * 16) & (Ln - 1)) + hsel * 8;
    v8f* acc = mi ? acc1 : acc0;
#pragma unroll
    for (int t = 0; t < 4; ++t) {
      const int n = n0 + t * 16 + row;
      const float bnv = bias[n];
      const int h = n >> 6, d = n & 63;
      if (mode == 2) {
        v8bf pk;
#pragma unroll
        for (int j = 0; j < 8; ++j) pk[j] = (__bf16)(acc[t][j] + bnv);
        bf16* dst = out + (((size_t)(bidx * Hn + h) * HDn + d) * Ln + l0);
        *(v8bf*)dst = pk;
      } else {
        const float sc = (mode == 0) ? QSCALE : 1.0f;
#pragma unroll
        for (int j = 0; j < 8; ++j) {
          out[((size_t)(bidx * Hn + h) * Ln + (l0 + j)) * HDn + d] =
              (__bf16)((acc[t][j] + bnv) * sc);
        }
      }
    }
  }
}

// ---------------------------------------------------------------- flash attention
// One wave per (bh, 16-query tile). Key loop unrolled x2; next step's K
// fragments prefetch during the softmax of the current step.
__global__ __launch_bounds__(32) void attn_kernel(
    const bf16* __restrict__ Q, const bf16* __restrict__ K,
    const bf16* __restrict__ Vt, const float* __restrict__ GW,
    bf16* __restrict__ O) {
  __shared__ __bf16 pbuf[16 * 32];
  const int lane = threadIdx.x & 31;
  const int row = lane & 15, hsel = lane >> 4;
  const int bh = blockIdx.y;
  const int q0 = blockIdx.x * 16;

  const bf16* qrow = Q + ((size_t)bh * Ln + q0 + row) * HDn;
  v16bf aQ[2];
#pragma unroll
  for (int c = 0; c < 2; ++c)
    aQ[c] = cat_bf(*(const v8bf*)(qrow + c * 32 + hsel * 8),
                   *(const v8bf*)(qrow + c * 32 + 16 + hsel * 8));

  const float* gwrow = GW + (size_t)(bh >> 1) * Ln;
  const bf16* kbase = K + (size_t)bh * Ln * HDn + (size_t)row * HDn;

  float Mx[8], Ls[8];
  v8f acc[4] = {v8f{}, v8f{}, v8f{}, v8f{}};
#pragma unroll
  for (int j = 0; j < 8; ++j) { Mx[j] = -3.0e38f; Ls[j] = 0.0f; }

  auto loadK = [&](int kt, v16bf (&bK)[2][2]) {
#pragma unroll
    for (int si = 0; si < 2; ++si) {
      const bf16* krow = kbase + (size_t)(kt + si * 16) * HDn;
#pragma unroll
      for (int ch = 0; ch < 2; ++ch)
        bK[si][ch] = cat_bf(*(const v8bf*)(krow + ch * 32 + hsel * 8),
                            *(const v8bf*)(krow + ch * 32 + 16 + hsel * 8));
    }
  };

  auto kstep = [&](int kt, v16bf (&bKc)[2][2], v16bf (&bKn)[2][2]) {
    // V fragments for this step: issued now, consumed by PV WMMAs at the end
    v16bf bV[4];
#pragma unroll
    for (int t = 0; t < 4; ++t) {
      const bf16* vrow = Vt + ((size_t)bh * HDn + t * 16 + row) * Ln + kt + hsel * 8;
      bV[t] = cat_bf(*(const v8bf*)vrow, *(const v8bf*)(vrow + 16));
    }
    // S = (Q K^T + bias) * gw
    v8f s[2];
#pragma unroll
    for (int si = 0; si < 2; ++si) {
      v8f c = v8f{};
      c = wmma_bf16(aQ[0], bKc[si][0], c);
      c = wmma_bf16(aQ[1], bKc[si][1], c);
      const float gwv = gwrow[kt + si * 16 + row];
#pragma unroll
      for (int j = 0; j < 8; ++j) c[j] = (c[j] + ATTN_BIAS) * gwv;
      s[si] = c;
    }
    // prefetch next step's K fragments; latency hides under the softmax below
    const int ktn = (kt + 32 < Ln) ? kt + 32 : 0;
    loadK(ktn, bKn);
    // online softmax (row j lives in VGPR j across the 16-lane half)
#pragma unroll
    for (int j = 0; j < 8; ++j) {
      float tm = fmaxf(s[0][j], s[1][j]);
#pragma unroll
      for (int mm = 1; mm < 16; mm <<= 1) tm = fmaxf(tm, __shfl_xor(tm, mm, 32));
      const float nm = fmaxf(Mx[j], tm);
      const float rescale = __expf(Mx[j] - nm);
      Ls[j] *= rescale;
#pragma unroll
      for (int t = 0; t < 4; ++t) acc[t][j] *= rescale;
      const float p0 = __expf(s[0][j] - nm);
      const float p1 = __expf(s[1][j] - nm);
      float rs = p0 + p1;
#pragma unroll
      for (int mm = 1; mm < 16; mm <<= 1) rs += __shfl_xor(rs, mm, 32);
      Ls[j] += rs;
      Mx[j] = nm;
      pbuf[(j + 8 * hsel) * 32 + row]      = (__bf16)p0;
      pbuf[(j + 8 * hsel) * 32 + 16 + row] = (__bf16)p1;
    }
    __syncthreads();
    // P tile back out of LDS in A-fragment layout
    const __bf16* prow = pbuf + row * 32;
    v16bf aP = cat_bf(*(const v8bf*)(prow + hsel * 8),
                      *(const v8bf*)(prow + 16 + hsel * 8));
    __syncthreads();
#pragma unroll
    for (int t = 0; t < 4; ++t) acc[t] = wmma_bf16(aP, bV[t], acc[t]);
  };

  v16bf bKA[2][2], bKB[2][2];
  loadK(0, bKA);
  for (int kt = 0; kt < Ln; kt += 64) {
    kstep(kt,      bKA, bKB);
    kstep(kt + 32, bKB, bKA);
  }

  const int b = bh >> 4, h = bh & 15;
  const int lbase = q0 + 8 * hsel;
#pragma unroll
  for (int t = 0; t < 4; ++t) {
    const int d = t * 16 + row;
#pragma unroll
    for (int j = 0; j < 8; ++j) {
      O[((size_t)b * Ln + (lbase + j)) * En + h * HDn + d] = (__bf16)(acc[t][j] / Ls[j]);
    }
  }
}

// ---------------------------------------------------------------- out-projection
__global__ __launch_bounds__(128) void oproj_kernel(
    const bf16* __restrict__ A, const bf16* __restrict__ W,
    const float* __restrict__ bias, float* __restrict__ out) {
  const int lane = threadIdx.x & 31;
  const int wv   = threadIdx.x >> 5;
  const int row  = lane & 15;
  const int hsel = lane >> 4;
  const int m0 = blockIdx.x * 128 + wv * 32;
  const int n0 = blockIdx.y * 64;

  const bf16* arow0 = A + (size_t)(m0 + row) * En;
  const bf16* arow1 = arow0 + (size_t)16 * En;
  const bf16* wrow[4];
#pragma unroll
  for (int t = 0; t < 4; ++t) wrow[t] = W + (size_t)(n0 + t * 16 + row) * En;

  v8f acc0[4] = {v8f{}, v8f{}, v8f{}, v8f{}};
  v8f acc1[4] = {v8f{}, v8f{}, v8f{}, v8f{}};

  v16bf a0A, a1A, bA[4], a0B, a1B, bB[4];

  auto loadset = [&](int k, v16bf& a0d, v16bf& a1d, v16bf (&bd)[4]) {
    a0d = load_afrag(arow0, k, hsel);
    a1d = load_afrag(arow1, k, hsel);
#pragma unroll
    for (int t = 0; t < 4; ++t) bd[t] = load_bfrag(wrow[t], k, hsel);
  };
  auto compute = [&](v16bf& a0c, v16bf& a1c, v16bf (&bc)[4]) {
#pragma unroll
    for (int t = 0; t < 4; ++t) {
      acc0[t] = wmma_bf16(a0c, bc[t], acc0[t]);
      acc1[t] = wmma_bf16(a1c, bc[t], acc1[t]);
    }
  };

  loadset(0, a0A, a1A, bA);
  for (int k = 0; k < En; k += 64) {
    loadset(k + 32, a0B, a1B, bB);
    compute(a0A, a1A, bA);
    const int kn = (k + 64 < En) ? k + 64 : 0;
    loadset(kn, a0A, a1A, bA);
    compute(a0B, a1B, bB);
  }

#pragma unroll
  for (int mi = 0; mi < 2; ++mi) {
    v8f* acc = mi ? acc1 : acc0;
#pragma unroll
    for (int t = 0; t < 4; ++t) {
      const int n = n0 + t * 16 + row;
      const float bnv = bias[n];
#pragma unroll
      for (int j = 0; j < 8; ++j) {
        out[(size_t)(m0 + mi * 16 + j + hsel * 8) * En + n] = acc[t][j] + bnv;
      }
    }
  }
}

// ---------------------------------------------------------------- launch
extern "C" void kernel_launch(void* const* d_in, const int* in_sizes, int n_in,
                              void* d_out, int out_size, void* d_ws, size_t ws_size,
                              hipStream_t stream) {
  (void)in_sizes; (void)n_in; (void)out_size; (void)ws_size;
  const float* q     = (const float*)d_in[0];
  const float* k     = (const float*)d_in[1];
  const float* v     = (const float*)d_in[2];
  const float* gauss = (const float*)d_in[3];
  const float* win   = (const float*)d_in[4];
  const float* bin   = (const float*)d_in[5];
  const float* wout  = (const float*)d_in[6];
  const float* bout  = (const float*)d_in[7];

  const size_t tElems = (size_t)Mn * En;
  bf16* qws   = (bf16*)d_ws;
  bf16* kws   = qws   + tElems;
  bf16* vtws  = kws   + tElems;
  bf16* aows  = vtws  + tElems;
  bf16* qin   = aows  + tElems;
  bf16* kin   = qin   + tElems;
  bf16* vin   = kin   + tElems;
  bf16* winbf = vin   + tElems;
  bf16* woutbf= winbf + (size_t)3 * En * En;
  float* gwb  = (float*)(woutbf + (size_t)En * En);

  const int n8t = (int)(tElems / 8), n8wi = 3 * En * En / 8, n8wo = En * En / 8;
  cvt_kernel<<<(n8t  + 255) / 256, 256, 0, stream>>>(q,    qin,    n8t);
  cvt_kernel<<<(n8t  + 255) / 256, 256, 0, stream>>>(k,    kin,    n8t);
  cvt_kernel<<<(n8t  + 255) / 256, 256, 0, stream>>>(v,    vin,    n8t);
  cvt_kernel<<<(n8wi + 255) / 256, 256, 0, stream>>>(win,  winbf,  n8wi);
  cvt_kernel<<<(n8wo + 255) / 256, 256, 0, stream>>>(wout, woutbf, n8wo);
  gw_kernel<<<((BHn / 2) * Ln + 255) / 256, 256, 0, stream>>>(gauss, gwb);

  dim3 pg(Mn / 128, En / 64), pb(128);
  proj_kernel<<<pg, pb, 0, stream>>>(qin, winbf,                       bin,          qws,  0);
  proj_kernel<<<pg, pb, 0, stream>>>(kin, winbf + (size_t)En * En,     bin + En,     kws,  1);
  proj_kernel<<<pg, pb, 0, stream>>>(vin, winbf + 2 * (size_t)En * En, bin + 2 * En, vtws, 2);

  attn_kernel<<<dim3(Ln / 16, BHn), 32, 0, stream>>>(qws, kws, vtws, gwb, aows);

  oproj_kernel<<<pg, pb, 0, stream>>>(aows, woutbf, bout, (float*)d_out);
}
